// CLVSAModel_71339406787072
// MI455X (gfx1250) — compile-verified, matrix operand order checked by reading, not run on hardware
//
#include <hip/hip_runtime.h>
#include <hip/hip_bf16.h>

#define B_  32
#define CH_ 16
#define D_  1024
#define TE  96
#define TD  48

typedef __attribute__((ext_vector_type(16))) __bf16 v16bf;
typedef __attribute__((ext_vector_type(8)))  __bf16 bf16x8;
typedef __attribute__((ext_vector_type(8)))  float  v8f;
typedef __attribute__((ext_vector_type(4)))  float  f32x4;

__device__ __forceinline__ float sigmoid_(float x) { return 1.0f / (1.0f + __expf(-x)); }

// ---------------------------------------------------------------------------
// Generic WMMA GEMM:  C[m*ldc+n] = act( sum_k A[m*K+k] * Bt[n*K+k] + bias[n] )
// A: f32 (M x K) dense row-major (converted to bf16 in-register).
// Bt: bf16 (Npad x K): pre-transposed weights, K-major, zero padded.
// K % 32 == 0, M % 16 == 0.  grid = (ceil(N/16), M/16), block = 32 (one wave).
// ---------------------------------------------------------------------------
__global__ void gemm_bf16_wmma(const float* __restrict__ A,
                               const __bf16* __restrict__ Bt,
                               float* __restrict__ C, int ldc,
                               const float* __restrict__ bias,
                               int N, int K, int act)
{
    const int lane = threadIdx.x;
    const int l15  = lane & 15;
    const int half = lane >> 4;                 // 0: lanes 0-15, 1: lanes 16-31
    const int tm   = blockIdx.y << 4;
    const int tn   = blockIdx.x << 4;

    const float*  Arow = A  + (size_t)(tm + l15) * K;
    const __bf16* Brow = Bt + (size_t)(tn + l15) * K;
    const int klo = half << 3;                  // elements 0..7  <- K = klo..klo+7
    const int khi = 16 + (half << 3);           // elements 8..15 <- K = khi..khi+7

    v8f acc = {0.f, 0.f, 0.f, 0.f, 0.f, 0.f, 0.f, 0.f};

    for (int k0 = 0; k0 < K; k0 += 32) {
        f32x4 a0 = *(const f32x4*)(Arow + k0 + klo);
        f32x4 a1 = *(const f32x4*)(Arow + k0 + klo + 4);
        f32x4 a2 = *(const f32x4*)(Arow + k0 + khi);
        f32x4 a3 = *(const f32x4*)(Arow + k0 + khi + 4);
        bf16x8 b0 = *(const bf16x8*)(Brow + k0 + klo);
        bf16x8 b1 = *(const bf16x8*)(Brow + k0 + khi);

        v16bf af, bfv;
#pragma unroll
        for (int j = 0; j < 4; ++j) {
            af[j]      = (__bf16)a0[j];
            af[4 + j]  = (__bf16)a1[j];
            af[8 + j]  = (__bf16)a2[j];
            af[12 + j] = (__bf16)a3[j];
        }
#pragma unroll
        for (int j = 0; j < 8; ++j) { bfv[j] = b0[j]; bfv[8 + j] = b1[j]; }

        acc = __builtin_amdgcn_wmma_f32_16x16x32_bf16(false, af, false, bfv,
                                                      (short)0, acc, false, false);
    }

    const int n = tn + l15;
    if (n < N) {
        const float bv = bias ? bias[n] : 0.f;
#pragma unroll
        for (int j = 0; j < 8; ++j) {
            float v = acc[j] + bv;
            if (act == 1)      v = tanhf(v);
            else if (act == 2) v = fmaxf(v, 0.f);
            const int m = tm + (half << 3) + j;
            C[(size_t)m * ldc + n] = v;
        }
    }
}

// ---------------------------------------------------------------------------
// Weight prep: f32 (K,N) -> bf16 (Npad,K) transposed, zero padded (K % 32 == 0)
// ---------------------------------------------------------------------------
__global__ void wcvt_trans(const float* __restrict__ W, __bf16* __restrict__ Wt,
                           int K, int N, int Npad)
{
    int idx = blockIdx.x * blockDim.x + threadIdx.x;
    int total = Npad * K;
    if (idx >= total) return;
    int n = idx / K, k = idx - n * K;
    float v = (n < N) ? W[(size_t)k * N + n] : 0.f;
    Wt[idx] = (__bf16)v;
}

// Conv weights: f32 (Nrows,K) flat -> bf16 (Nrows,Kpad) zero padded
__global__ void wcvt_rows(const float* __restrict__ W, __bf16* __restrict__ Wt,
                          int K, int Kpad, int Nrows)
{
    int idx = blockIdx.x * blockDim.x + threadIdx.x;
    int total = Nrows * Kpad;
    if (idx >= total) return;
    int n = idx / Kpad, k = idx - n * Kpad;
    Wt[idx] = (__bf16)((k < K) ? W[(size_t)n * K + k] : 0.f);
}

__global__ void zero_f32(float* __restrict__ p, int n)
{
    int i = blockIdx.x * blockDim.x + threadIdx.x;
    if (i < n) p[i] = 0.f;
}

// ---------------------------------------------------------------------------
// im2col for 3x3 SAME conv over concat([x (Cin ch), h (16 ch)]), 8x8 spatial.
// patch row m = b*64 + r*8 + c ; col k = ch*9 + (dr+1)*3 + (dc+1); pad->Kpad 0s
// ---------------------------------------------------------------------------
__global__ void im2col_kernel(const float* __restrict__ x, long xStrideB, int Cin,
                              const float* __restrict__ h,
                              float* __restrict__ patch, int K, int Kpad)
{
    int idx = blockIdx.x * blockDim.x + threadIdx.x;
    int total = B_ * 64 * Kpad;
    if (idx >= total) return;
    int m = idx / Kpad, k = idx - m * Kpad;
    float v = 0.f;
    if (k < K) {
        int b = m >> 6, rc = m & 63;
        int r = rc >> 3, c = rc & 7;
        int ch = k / 9, rem = k - ch * 9;
        int rr = r + rem / 3 - 1, cc = c + rem % 3 - 1;
        if (rr >= 0 && rr < 8 && cc >= 0 && cc < 8) {
            if (ch < Cin) v = x[(size_t)b * xStrideB + ch * 64 + rr * 8 + cc];
            else          v = h[(size_t)b * (CH_ * 64) + (ch - Cin) * 64 + rr * 8 + cc];
        }
    }
    patch[idx] = v;
}

// ---------------------------------------------------------------------------
// LSTM gates: G (B*64 spatial, 64 gates) -> update h,c (B, CH*64); stash h->raw
// gate order i,f,g,o over channel blocks of 16.
// ---------------------------------------------------------------------------
__global__ void lstm_gates(const float* __restrict__ G,
                           float* __restrict__ h, float* __restrict__ c,
                           float* __restrict__ raw, long rawStrideB)
{
    int idx = blockIdx.x * blockDim.x + threadIdx.x;
    if (idx >= B_ * D_) return;
    int b = idx >> 10, d = idx & 1023;
    int ch = d >> 6, rc = d & 63;
    const float* g = G + (size_t)(b * 64 + rc) * 64;
    float gi = g[ch], gf = g[CH_ + ch], gg = g[2 * CH_ + ch], go = g[3 * CH_ + ch];
    float cn = sigmoid_(gf) * c[idx] + sigmoid_(gi) * tanhf(gg);
    float hv = sigmoid_(go) * tanhf(cn);
    c[idx] = cn;
    h[idx] = hv;
    if (raw) raw[(size_t)b * rawStrideB + d] = hv;
}

// ---------------------------------------------------------------------------
// Attention scores + softmax + context, fused per batch row. block=256 (8 waves)
// Writes Acat[b] = [ q (1024) | ctx (1024) ] for the WMMA projection GEMM.
// ---------------------------------------------------------------------------
__global__ void attend_kernel(const float* __restrict__ q, long qStrideB,
                              const float* __restrict__ Kb, const float* __restrict__ Vb,
                              long kvStrideB, int tvalid,
                              float* __restrict__ Acat)
{
    __shared__ float sc[128];
    const int b = blockIdx.x, tid = threadIdx.x;
    const int wave = tid >> 5, lane = tid & 31;
    const float* qb  = q  + (size_t)b * qStrideB;
    const float* Kbb = Kb + (size_t)b * kvStrideB;

    for (int t = wave; t < tvalid; t += 8) {
        const float* kt = Kbb + (size_t)t * D_;
        float s = 0.f;
        for (int d = lane; d < D_; d += 32) s += qb[d] * kt[d];
#pragma unroll
        for (int off = 16; off > 0; off >>= 1) s += __shfl_down(s, off, 32);
        if (lane == 0) sc[t] = s * 0.03125f;   // 1/sqrt(1024)
    }
    __syncthreads();

    if (wave == 0) {
        float m = -1e30f;
        for (int t = lane; t < tvalid; t += 32) m = fmaxf(m, sc[t]);
#pragma unroll
        for (int off = 16; off > 0; off >>= 1) m = fmaxf(m, __shfl_xor(m, off, 32));
        float s = 0.f;
        for (int t = lane; t < tvalid; t += 32) { float e = __expf(sc[t] - m); sc[t] = e; s += e; }
#pragma unroll
        for (int off = 16; off > 0; off >>= 1) s += __shfl_xor(s, off, 32);
        float inv = 1.f / s;
        for (int t = lane; t < tvalid; t += 32) sc[t] *= inv;
    }
    __syncthreads();

    const float* Vbb = Vb + (size_t)b * kvStrideB;
    for (int d = tid; d < D_; d += 256) {
        float a = 0.f;
        for (int t = 0; t < tvalid; ++t) a += sc[t] * Vbb[(size_t)t * D_ + d];
        Acat[(size_t)b * 2048 + 1024 + d] = a;
        Acat[(size_t)b * 2048 + d]        = qb[d];
    }
}

__global__ void copy_rows(const float* __restrict__ src, long sStrideB,
                          float* __restrict__ dst, long dStrideB, int rowLen)
{
    int idx = blockIdx.x * blockDim.x + threadIdx.x;
    if (idx >= B_ * rowLen) return;
    int b = idx / rowLen, j = idx - b * rowLen;
    dst[(size_t)b * dStrideB + j] = src[(size_t)b * sStrideB + j];
}

// Posterior input: time-flipped concat of x_dec and label embedding -> (B,TD,2,64)
__global__ void build_xpost(const float* __restrict__ x_dec, const int* __restrict__ y_dec,
                            const float* __restrict__ emb, float* __restrict__ xpost)
{
    int idx = blockIdx.x * blockDim.x + threadIdx.x;
    if (idx >= B_ * TD * 64) return;
    int b = idx / (TD * 64); int rem = idx - b * TD * 64;
    int t = rem / 64, rc = rem - t * 64;
    int trev = TD - 1 - t;
    xpost[((size_t)(b * TD + t) * 2 + 0) * 64 + rc] = x_dec[((size_t)b * TD + trev) * 64 + rc];
    int y = y_dec[b * TD + trev];
    xpost[((size_t)(b * TD + t) * 2 + 1) * 64 + rc] = emb[y * 64 + rc];
}

// ---------------------------------------------------------------------------
// Host orchestration
// ---------------------------------------------------------------------------
enum {
    IN_XENC = 0, IN_XDEC = 1, IN_YDEC = 2,
    P_ENC_C0W = 3,  P_ENC_C0B = 4,  P_ENC_C1W = 5,  P_ENC_C1B = 6,
    P_ENC_A0W = 7,  P_ENC_A0B = 8,  P_ENC_A1W = 9,  P_ENC_A1B = 10,
    P_DEC_C0W = 11, P_DEC_C0B = 12, P_DEC_C1W = 13, P_DEC_C1B = 14,
    P_INT_A0W = 15, P_INT_A0B = 16, P_INT_A1W = 17, P_INT_A1B = 18,
    P_SLF_A0W = 19, P_SLF_A0B = 20, P_SLF_A1W = 21, P_SLF_A1B = 22,
    P_BWD_C0W = 23, P_BWD_C0B = 24, P_BWD_C1W = 25, P_BWD_C1B = 26,
    P_BWD_A0W = 27, P_BWD_A0B = 28, P_BWD_A1W = 29, P_BWD_A1B = 30,
    P_EMB = 31,
    P_PRI_NW = 32, P_PRI_NB = 33, P_PRI_MW = 34, P_PRI_MB = 35, P_PRI_LW = 36, P_PRI_LB = 37,
    P_POS_NW = 38, P_POS_NB = 39, P_POS_MW = 40, P_POS_MB = 41, P_POS_LW = 42, P_POS_LB = 43,
    P_HEAD_W = 44, P_HEAD_B = 45, P_BHD_W = 46,  P_BHD_B = 47
};

extern "C" void kernel_launch(void* const* d_in, const int* in_sizes, int n_in,
                              void* d_out, int out_size, void* d_ws, size_t ws_size,
                              hipStream_t stream)
{
    (void)in_sizes; (void)n_in; (void)out_size; (void)ws_size;
    auto F = [&](int i) { return (const float*)d_in[i]; };
    float* outp = (float*)d_out;

    // Output slots (logits, p_mu, p_lv, q_mu, q_lv, bwd_logits), flat concat.
    float* o_logits = outp;                       // (B,48,3)
    float* o_pmu    = outp + 4608;                // (B,48,32)
    float* o_plv    = outp + 4608 + 49152;
    float* o_qmu    = outp + 4608 + 2 * 49152;
    float* o_qlv    = outp + 4608 + 3 * 49152;
    float* o_blog   = outp + 4608 + 4 * 49152;    // (B,48,3)

    // Workspace bump allocator (256B aligned)
    char* wbase = (char*)d_ws;
    size_t woff = 0;
    auto alloc  = [&](size_t bytes) -> char* {
        char* p = wbase + woff;
        woff = (woff + bytes + 255) & ~(size_t)255;
        return p;
    };
    auto allocF = [&](size_t n) { return (float*)alloc(n * sizeof(float)); };
    auto allocH = [&](size_t n) { return (__bf16*)alloc(n * sizeof(__bf16)); };

    // Zeroed state region: h/c for enc(+dec) and posterior, z ping-pong.
    const int zn = 8 * B_ * D_ + 2 * B_ * 32;
    float* zr = allocF(zn);
    float* enc_h[2]  = { zr + 0 * B_ * D_, zr + 1 * B_ * D_ };
    float* enc_c[2]  = { zr + 2 * B_ * D_, zr + 3 * B_ * D_ };
    float* post_h[2] = { zr + 4 * B_ * D_, zr + 5 * B_ * D_ };
    float* post_c[2] = { zr + 6 * B_ * D_, zr + 7 * B_ * D_ };
    float* zb0 = zr + 8 * B_ * D_;
    float* zb1 = zb0 + B_ * 32;
    zero_f32<<<(zn + 255) / 256, 256, 0, stream>>>(zr, zn);

    // Weight conversion (once per call; graph replays include it — weights constant).
    auto cvtT = [&](const float* W, int K, int N, int Npad) -> __bf16* {
        __bf16* p = allocH((size_t)Npad * K);
        int total = Npad * K;
        wcvt_trans<<<(total + 255) / 256, 256, 0, stream>>>(W, p, K, N, Npad);
        return p;
    };
    auto cvtC = [&](const float* W, int K) -> __bf16* {
        int Kpad = (K + 31) & ~31;
        __bf16* p = allocH((size_t)64 * Kpad);
        int total = 64 * Kpad;
        wcvt_rows<<<(total + 255) / 256, 256, 0, stream>>>(W, p, K, Kpad, 64);
        return p;
    };
    __bf16* w_encA[2] = { cvtT(F(P_ENC_A0W), 2048, 1024, 1024), cvtT(F(P_ENC_A1W), 2048, 1024, 1024) };
    __bf16* w_intA[2] = { cvtT(F(P_INT_A0W), 2048, 1024, 1024), cvtT(F(P_INT_A1W), 2048, 1024, 1024) };
    __bf16* w_slfA[2] = { cvtT(F(P_SLF_A0W), 2048, 1024, 1024), cvtT(F(P_SLF_A1W), 2048, 1024, 1024) };
    __bf16* w_bwdA[2] = { cvtT(F(P_BWD_A0W), 2048, 1024, 1024), cvtT(F(P_BWD_A1W), 2048, 1024, 1024) };
    __bf16* w_encC[2] = { cvtC(F(P_ENC_C0W), 17 * 9), cvtC(F(P_ENC_C1W), 32 * 9) };
    __bf16* w_decC[2] = { cvtC(F(P_DEC_C0W), 17 * 9), cvtC(F(P_DEC_C1W), 32 * 9) };
    __bf16* w_bwdC[2] = { cvtC(F(P_BWD_C0W), 18 * 9), cvtC(F(P_BWD_C1W), 32 * 9) };
    __bf16* w_priN = cvtT(F(P_PRI_NW), 1056, 512, 512);
    __bf16* w_priM = cvtT(F(P_PRI_MW), 512, 32, 32);
    __bf16* w_priL = cvtT(F(P_PRI_LW), 512, 32, 32);
    __bf16* w_posN = cvtT(F(P_POS_NW), 1024, 256, 256);
    __bf16* w_posM = cvtT(F(P_POS_MW), 256, 32, 32);
    __bf16* w_posL = cvtT(F(P_POS_LW), 256, 32, 32);
    __bf16* w_head = cvtT(F(P_HEAD_W), 1056, 3, 16);
    __bf16* w_bhd  = cvtT(F(P_BHD_W), 1024, 3, 16);

    // Activation / history buffers
    float* raws_enc[2] = { allocF((size_t)B_ * TE * D_), allocF((size_t)B_ * TE * D_) };
    float* refs_enc[2] = { allocF((size_t)B_ * TE * D_), allocF((size_t)B_ * TE * D_) };
    float* raws_pst[2] = { allocF((size_t)B_ * TD * D_), allocF((size_t)B_ * TD * D_) };
    float* refs_pst[2] = { allocF((size_t)B_ * TD * D_), allocF((size_t)B_ * TD * D_) };
    float* pres_dec[2] = { allocF((size_t)B_ * TD * D_), allocF((size_t)B_ * TD * D_) };
    float* refs_dec[2] = { allocF((size_t)B_ * TD * D_), allocF((size_t)B_ * TD * D_) };
    float* patch = allocF((size_t)2048 * 288);
    float* Gbuf  = allocF((size_t)2048 * 64);
    float* Acat  = allocF((size_t)B_ * 2048);
    float* Acat2 = allocF((size_t)B_ * 1056);
    float* hvrD  = allocF((size_t)B_ * D_);
    float* preD  = allocF((size_t)B_ * D_);
    float* ph512 = allocF((size_t)B_ * 512);
    float* ph256 = allocF((size_t)B_ * 256);
    float* xpost = allocF((size_t)B_ * TD * 128);

    build_xpost<<<(B_ * TD * 64 + 255) / 256, 256, 0, stream>>>(
        F(IN_XDEC), (const int*)d_in[IN_YDEC], F(P_EMB), xpost);

    // Helpers
    auto gemm = [&](const float* A, const __bf16* Bt, float* C, int ldc, const float* bias,
                    int M, int N, int K, int act) {
        dim3 g((unsigned)((N + 15) / 16), (unsigned)(M / 16));
        gemm_bf16_wmma<<<g, 32, 0, stream>>>(A, Bt, C, ldc, bias, N, K, act);
    };
    auto copyR = [&](const float* s, long ss, float* d, long ds, int len) {
        copy_rows<<<(B_ * len + 255) / 256, 256, 0, stream>>>(s, ss, d, ds, len);
    };
    auto attendP = [&](const float* q, const float* Kb, const float* Vb, long kvS, int tv,
                       const __bf16* Wt, const float* bias, float* outD) {
        attend_kernel<<<B_, 256, 0, stream>>>(q, (long)D_, Kb, Vb, kvS, tv, Acat);
        gemm(Acat, Wt, outD, D_, bias, B_, D_, 2 * D_, 1 /*tanh*/);
    };
    auto convStep = [&](const float* x, long xs, int Cin, const __bf16* Wt, const float* bias,
                        float* h, float* c, float* raw, long rawS) {
        int K = (Cin + CH_) * 9, Kpad = (K + 31) & ~31;
        int tot = B_ * 64 * Kpad;
        im2col_kernel<<<(tot + 255) / 256, 256, 0, stream>>>(x, xs, Cin, h, patch, K, Kpad);
        gemm(patch, Wt, Gbuf, 64, bias, B_ * 64, 64, Kpad, 0);
        lstm_gates<<<(B_ * D_ + 255) / 256, 256, 0, stream>>>(Gbuf, h, c, raw, rawS);
    };

    // ------------------------- Posterior (backward ConvLSTM) -----------------
    for (int t = 0; t < TD; ++t) {
        const float* cur = nullptr;
        for (int li = 0; li < 2; ++li) {
            const float* x; long xs; int Cin;
            if (li == 0) { x = xpost + (size_t)t * 128; xs = (long)TD * 128; Cin = 2; }
            else         { x = cur;                     xs = D_;             Cin = CH_; }
            convStep(x, xs, Cin, w_bwdC[li], F(li ? P_BWD_C1B : P_BWD_C0B),
                     post_h[li], post_c[li], raws_pst[li] + (size_t)t * D_, (long)TD * D_);
            if (t == 0) {
                copyR(post_h[li], D_, refs_pst[li] + (size_t)t * D_, (long)TD * D_, D_);
                cur = post_h[li];
            } else {
                attendP(post_h[li], raws_pst[li], refs_pst[li], (long)TD * D_, t,
                        w_bwdA[li], F(li ? P_BWD_A1B : P_BWD_A0B), hvrD);
                copyR(hvrD, D_, refs_pst[li] + (size_t)t * D_, (long)TD * D_, D_);
                cur = hvrD;
            }
        }
        int trev = TD - 1 - t;  // un-flip time for outputs
        gemm(cur, w_posN, ph256, 256, F(P_POS_NB), B_, 256, 1024, 2 /*relu*/);
        gemm(ph256, w_posM, o_qmu + (size_t)trev * 32, TD * 32, F(P_POS_MB), B_, 32, 256, 0);
        gemm(ph256, w_posL, o_qlv + (size_t)trev * 32, TD * 32, F(P_POS_LB), B_, 32, 256, 0);
        gemm(cur, w_bhd, o_blog + (size_t)trev * 3, TD * 3, F(P_BHD_B), B_, 3, 1024, 0);
    }

    // ------------------------------- Encoder ---------------------------------
    for (int t = 0; t < TE; ++t) {
        const float* cur = nullptr;
        for (int li = 0; li < 2; ++li) {
            const float* x; long xs; int Cin;
            if (li == 0) { x = F(IN_XENC) + (size_t)t * 64; xs = (long)TE * 64; Cin = 1; }
            else         { x = cur;                         xs = D_;            Cin = CH_; }
            convStep(x, xs, Cin, w_encC[li], F(li ? P_ENC_C1B : P_ENC_C0B),
                     enc_h[li], enc_c[li], raws_enc[li] + (size_t)t * D_, (long)TE * D_);
            if (t == 0) {
                copyR(enc_h[li], D_, refs_enc[li] + (size_t)t * D_, (long)TE * D_, D_);
                cur = enc_h[li];
            } else {
                attendP(enc_h[li], raws_enc[li], refs_enc[li], (long)TE * D_, t,
                        w_encA[li], F(li ? P_ENC_A1B : P_ENC_A0B), hvrD);
                copyR(hvrD, D_, refs_enc[li] + (size_t)t * D_, (long)TE * D_, D_);
                cur = hvrD;
            }
        }
    }

    // ------------------------------- Decoder ---------------------------------
    float* zprev = zb0; float* zcur = zb1;
    const float* enc_out = refs_enc[1];
    for (int t = 0; t < TD; ++t) {
        const float* cur = nullptr;
        for (int li = 0; li < 2; ++li) {
            const float* x; long xs; int Cin;
            if (li == 0) { x = F(IN_XDEC) + (size_t)t * 64; xs = (long)TD * 64; Cin = 1; }
            else         { x = cur;                         xs = D_;            Cin = CH_; }
            convStep(x, xs, Cin, w_decC[li], F(li ? P_DEC_C1B : P_DEC_C0B),
                     enc_h[li], enc_c[li], nullptr, 0);
            // inter-attention over full encoder outputs
            attendP(enc_h[li], enc_out, enc_out, (long)TE * D_, TE,
                    w_intA[li], F(li ? P_INT_A1B : P_INT_A0B), preD);
            copyR(preD, D_, pres_dec[li] + (size_t)t * D_, (long)TD * D_, D_);
            if (t == 0) {
                copyR(preD, D_, refs_dec[li] + (size_t)t * D_, (long)TD * D_, D_);
                cur = preD;
            } else {
                attendP(preD, pres_dec[li], refs_dec[li], (long)TD * D_, t,
                        w_slfA[li], F(li ? P_SLF_A1B : P_SLF_A0B), hvrD);
                copyR(hvrD, D_, refs_dec[li] + (size_t)t * D_, (long)TD * D_, D_);
                cur = hvrD;
            }
        }
        // prior: concat [hv, prev_z] -> relu FC -> mu/logvar; z_t = p_mu
        copyR(cur, D_, Acat2, 1056, D_);
        copyR(zprev, 32, Acat2 + 1024, 1056, 32);
        gemm(Acat2, w_priN, ph512, 512, F(P_PRI_NB), B_, 512, 1056, 2 /*relu*/);
        gemm(ph512, w_priM, zcur, 32, F(P_PRI_MB), B_, 32, 512, 0);
        copyR(zcur, 32, o_pmu + (size_t)t * 32, TD * 32, 32);
        gemm(ph512, w_priL, o_plv + (size_t)t * 32, TD * 32, F(P_PRI_LB), B_, 32, 512, 0);
        // head: concat [hv, z_t] -> logits
        copyR(cur, D_, Acat2, 1056, D_);
        copyR(zcur, 32, Acat2 + 1024, 1056, 32);
        gemm(Acat2, w_head, o_logits + (size_t)t * 3, TD * 3, F(P_HEAD_B), B_, 3, 1056, 0);
        float* tmp = zprev; zprev = zcur; zcur = tmp;
    }
}
